// ConvBlockFD_10144712753919
// MI455X (gfx1250) — matrix operand order repaired and checked by reading, not compiled, not run on hardware
//
#include <hip/hip_runtime.h>
#include <hip/hip_bf16.h>
#include <stdint.h>

// ---------------- problem constants ----------------
#define B_   16
#define C1_  128     // Cin  of layer 1
#define C2_  256     // Cout of both layers (also Cin of layer 2)
#define H_   128
#define W_   128
#define CO_  256
#define HW_  (H_*W_)

typedef __bf16 bf16;
typedef __attribute__((ext_vector_type(16))) __bf16 v16bf;
typedef __attribute__((ext_vector_type(8)))  __bf16 v8bf;
typedef __attribute__((ext_vector_type(8)))  float  v8f;

// ---------------- ws layout (bytes, 256-aligned) ----------------
#define OFF_XH    ((size_t)0)                         //  16*128*128*128 bf16 = 64 MB
#define OFF_Y1H   ((size_t)67108864)                  //  16*256*128*128 bf16 = 128 MB
#define OFF_WD1   ((size_t)(67108864+134217728))      //  16*9*256*128 bf16
#define OFF_WD2   (OFF_WD1 + (size_t)9437184)         //  16*9*256*256 bf16
#define OFF_GAP1  (OFF_WD2 + (size_t)18874368)
#define OFF_GAP2  (OFF_GAP1 + (size_t)8192)
#define OFF_AT1   (OFF_GAP2 + (size_t)16384)
#define OFF_AT2   (OFF_AT1  + (size_t)256)

// ---------------- small helper kernels ----------------
__global__ void k_f32_to_bf16(const float* __restrict__ in, bf16* __restrict__ out, int n) {
    int i = blockIdx.x * 256 + threadIdx.x;
    if (i < n) out[i] = (bf16)in[i];
}

__global__ void k_gap_f32(const float* __restrict__ x, float* __restrict__ gap) {
    const float* p = x + (size_t)blockIdx.x * HW_;
    float s = 0.f;
    for (int i = threadIdx.x; i < HW_; i += 256) s += p[i];
    __shared__ float red[256];
    red[threadIdx.x] = s; __syncthreads();
    for (int st = 128; st > 0; st >>= 1) {
        if (threadIdx.x < st) red[threadIdx.x] += red[threadIdx.x + st];
        __syncthreads();
    }
    if (threadIdx.x == 0) gap[blockIdx.x] = red[0] * (1.f / HW_);
}

__global__ void k_gap_bf16(const bf16* __restrict__ x, float* __restrict__ gap) {
    const bf16* p = x + (size_t)blockIdx.x * HW_;
    float s = 0.f;
    for (int i = threadIdx.x; i < HW_; i += 256) s += (float)p[i];
    __shared__ float red[256];
    red[threadIdx.x] = s; __syncthreads();
    for (int st = 128; st > 0; st >>= 1) {
        if (threadIdx.x < st) red[threadIdx.x] += red[threadIdx.x + st];
        __syncthreads();
    }
    if (threadIdx.x == 0) gap[blockIdx.x] = red[0] * (1.f / HW_);
}

// GAP[B,C] -> relu(gap@w1+b1) -> softmax(h@w2+b2) -> attn[B,4]. One block, thread=sample.
__global__ void k_attn(const float* __restrict__ gap, const float* __restrict__ w1,
                       const float* __restrict__ b1, const float* __restrict__ w2,
                       const float* __restrict__ b2, float* __restrict__ attn,
                       int C, int Hd) {
    int b = threadIdx.x;
    if (b >= B_) return;
    float h[64];
    for (int j = 0; j < Hd; ++j) {
        float s = b1[j];
        for (int c = 0; c < C; ++c) s += gap[b * C + c] * w1[c * Hd + j];
        h[j] = s > 0.f ? s : 0.f;
    }
    float lg[4]; float mx = -1e30f;
    for (int k = 0; k < 4; ++k) {
        float s = b2[k];
        for (int j = 0; j < Hd; ++j) s += h[j] * w2[j * 4 + k];
        lg[k] = s; mx = fmaxf(mx, s);
    }
    float den = 0.f;
    for (int k = 0; k < 4; ++k) { lg[k] = __expf(lg[k] - mx); den += lg[k]; }
    for (int k = 0; k < 4; ++k) attn[b * 4 + k] = lg[k] / den;
}

// Fuse attention-mix (linear) with irfft2(s=(3,3)).
// irfft2 = ifft(axis=-2) then irfft(axis=-1, n=3) which discards Im(DC bin).
// Output layout: wd[b][tap(kh*3+kw)][co][ci] in bf16 (WMMA A-friendly: ci contiguous).
__global__ void k_mix_irfft(const float* __restrict__ wfr, const float* __restrict__ wfi,
                            const float* __restrict__ attn, bf16* __restrict__ wd,
                            int Ci) {
    long long idx = (long long)blockIdx.x * 256 + threadIdx.x;
    long long tot = (long long)B_ * CO_ * Ci;
    if (idx >= tot) return;
    int ci = (int)(idx % Ci);
    int co = (int)((idx / Ci) % CO_);
    int b  = (int)(idx / ((long long)Ci * CO_));

    float Fr[6], Fi[6];
    #pragma unroll
    for (int t = 0; t < 6; ++t) { Fr[t] = 0.f; Fi[t] = 0.f; }
    #pragma unroll
    for (int k = 0; k < 4; ++k) {
        float a = attn[b * 4 + k];
        const float* pr = wfr + (((size_t)k * CO_ + co) * Ci + ci) * 6;
        const float* pi = wfi + (((size_t)k * CO_ + co) * Ci + ci) * 6;
        #pragma unroll
        for (int t = 0; t < 6; ++t) { Fr[t] += a * pr[t]; Fi[t] += a * pi[t]; }
    }
    const float c3[3] = { 1.f, -0.5f, -0.5f };
    const float s3[3] = { 0.f, 0.86602540378443865f, -0.86602540378443865f };
    #pragma unroll
    for (int p = 0; p < 3; ++p) {          // row index (kh), ifft along axis -2 (e^{+i})
        float Gr0 = 0.f, Gr1 = 0.f, Gi1 = 0.f;
        #pragma unroll
        for (int u = 0; u < 3; ++u) {
            int t = (u * p) % 3; float c = c3[t], s = s3[t];
            Gr0 += Fr[u * 2 + 0] * c - Fi[u * 2 + 0] * s;   // Im of v=0 bin is discarded by irfft
            Gr1 += Fr[u * 2 + 1] * c - Fi[u * 2 + 1] * s;
            Gi1 += Fr[u * 2 + 1] * s + Fi[u * 2 + 1] * c;
        }
        #pragma unroll
        for (int n = 0; n < 3; ++n) {      // col index (kw), irfft n=3
            float v = (Gr0 + 2.f * (Gr1 * c3[n] - Gi1 * s3[n])) * (1.f / 9.f);
            wd[(((size_t)b * 9 + (p * 3 + n)) * CO_ + co) * Ci + ci] = (bf16)v;
        }
    }
}

// ---------------- WMMA conv: 9 shifted GEMMs, bf16 x bf16 -> f32 ----------------
// Block: 256 thr = 8 waves; tile Co=128 x full image row (128 px).
// wave mw=wave&3 -> M rows [co0+mw*32,+32) (2 frags); nw=wave>>2 -> cols [nw*64,+64) (4 frags).
// grid: x = co_tile(2); y = h; z = b.
__device__ inline v16bf load_a_frag(const bf16* p) {
    // 16-bit A 16x32 layout: lane<16 holds K 0..7 (elts 0-7) and 16..23 (elts 8-15);
    // lane>=16 holds K 8..15 and 24..31 -> two contiguous 16B chunks at +0 and +16 elems.
    v8bf lo = *(const v8bf*)p;
    v8bf hi = *(const v8bf*)(p + 16);
    v16bf r;
    #pragma unroll
    for (int t = 0; t < 8; ++t) { r[t] = lo[t]; r[t + 8] = hi[t]; }
    return r;
}

__global__ __launch_bounds__(256)
void k_conv_wmma(const bf16* __restrict__ xh, const bf16* __restrict__ wd,
                 const float* __restrict__ bias, bf16* __restrict__ outh,
                 float* __restrict__ outf, int Ci) {
    __shared__ __attribute__((aligned(32))) bf16 Xt[3][130][32];  // [row dh][w=-1+j][ci]

    const int tid  = threadIdx.x;
    const int lane = tid & 31;
    const int wave = tid >> 5;
    const int lm   = lane & 15;   // N col (B/C) or M row (A) within 16
    const int lh   = lane >> 4;   // half selector

    const int co0  = blockIdx.x * 128;
    const int h    = blockIdx.y;
    const int b    = blockIdx.z;

    const int rowBase = co0 + (wave & 3) * 32;
    const int colBase = (wave >> 2) * 64;

    const v8f vzero = {0.f,0.f,0.f,0.f,0.f,0.f,0.f,0.f};
    v8f acc[2][4];
    #pragma unroll
    for (int i = 0; i < 2; ++i)
        #pragma unroll
        for (int j = 0; j < 4; ++j) acc[i][j] = vzero;

    // per-lane weight base: wd[b][tap][rowBase+lm + i*16][kc*32 + lh*8 ..]
    const size_t tapStride = (size_t)CO_ * Ci;
    const bf16* wBase = wd + ((size_t)b * 9 * CO_ + rowBase + lm) * Ci + lh * 8;

    const int ciChunks = Ci >> 5;
    for (int kc = 0; kc < ciChunks; ++kc) {
        __syncthreads();
        // stage transposed tile: Xt[dh][j][c] = x[b][kc*32+c][h+dh-1][-1+j], zero-padded
        for (int t = tid; t < 3 * 32 * 130; t += 256) {
            int j   = t % 130;
            int rem = t / 130;
            int c   = rem & 31;
            int dh  = rem >> 5;
            int hs  = h + dh - 1;
            int wsc = j - 1;
            bf16 v = (bf16)0.f;
            if (hs >= 0 && hs < H_ && wsc >= 0 && wsc < W_)
                v = xh[(((size_t)b * Ci + (kc * 32 + c)) * H_ + hs) * W_ + wsc];
            Xt[dh][j][c] = v;
        }
        __syncthreads();

        const bf16* wKc = wBase + (size_t)kc * 32;
        if (kc + 1 < ciChunks) {
            // warm WGP$ with next chunk's weights (global_prefetch_b8)
            __builtin_prefetch(wKc + 32, 0, 1);
            __builtin_prefetch(wKc + 32 + 16 * (size_t)Ci, 0, 1);
        }

        #pragma unroll
        for (int tap = 0; tap < 9; ++tap) {
            const int kh = tap / 3, kw = tap % 3;
            // A fragments (weights), L2-resident
            const bf16* wp = wKc + tap * tapStride;
            v16bf af[2];
            af[0] = load_a_frag(wp);
            af[1] = load_a_frag(wp + (size_t)16 * Ci);
            // B fragments (activations) from LDS: lane<16 -> K 0..7,16..23, lane>=16 -> K 8..15,24..31
            v16bf bfv[4];
            #pragma unroll
            for (int j = 0; j < 4; ++j) {
                int col = colBase + j * 16 + lm;
                bfv[j] = load_a_frag(&Xt[kh][col + kw][lh * 8]);
            }
            #pragma unroll
            for (int i = 0; i < 2; ++i)
                #pragma unroll
                for (int j = 0; j < 4; ++j)
                    acc[i][j] = __builtin_amdgcn_wmma_f32_16x16x32_bf16(
                        false, af[i], false, bfv[j], (short)0, acc[i][j], false, false);
        }
    }

    // epilogue: C layout -> lane<16: N=lm, M=g ; lane>=16: N=lm, M=8+g
    #pragma unroll
    for (int i = 0; i < 2; ++i) {
        #pragma unroll
        for (int j = 0; j < 4; ++j) {
            #pragma unroll
            for (int g = 0; g < 8; ++g) {
                int row = rowBase + i * 16 + lh * 8 + g;
                int cl  = colBase + j * 16 + lm;
                float v = acc[i][j][g] + bias[row];
                v = v > 0.f ? v : 0.f;
                size_t o = (((size_t)b * CO_ + row) * H_ + h) * W_ + cl;
                if (outf) outf[o] = v;
                else      outh[o] = (bf16)v;
            }
        }
    }
}

// ---------------- driver ----------------
extern "C" void kernel_launch(void* const* d_in, const int* in_sizes, int n_in,
                              void* d_out, int out_size, void* d_ws, size_t ws_size,
                              hipStream_t stream) {
    const float* x     = (const float*)d_in[0];
    const float* w1_fr = (const float*)d_in[1];
    const float* w1_fi = (const float*)d_in[2];
    const float* b1    = (const float*)d_in[3];
    const float* a1w1  = (const float*)d_in[4];
    const float* a1b1  = (const float*)d_in[5];
    const float* a1w2  = (const float*)d_in[6];
    const float* a1b2  = (const float*)d_in[7];
    const float* w2_fr = (const float*)d_in[8];
    const float* w2_fi = (const float*)d_in[9];
    const float* b2    = (const float*)d_in[10];
    const float* a2w1  = (const float*)d_in[11];
    const float* a2b1  = (const float*)d_in[12];
    const float* a2w2  = (const float*)d_in[13];
    const float* a2b2  = (const float*)d_in[14];

    char* ws = (char*)d_ws;
    bf16*  xh    = (bf16*)(ws + OFF_XH);
    bf16*  y1h   = (bf16*)(ws + OFF_Y1H);
    bf16*  wd1   = (bf16*)(ws + OFF_WD1);
    bf16*  wd2   = (bf16*)(ws + OFF_WD2);
    float* gap1  = (float*)(ws + OFF_GAP1);
    float* gap2  = (float*)(ws + OFF_GAP2);
    float* attn1 = (float*)(ws + OFF_AT1);
    float* attn2 = (float*)(ws + OFF_AT2);

    // ---- layer 1 ----
    {
        int n = B_ * C1_ * HW_;
        k_f32_to_bf16<<<(n + 255) / 256, 256, 0, stream>>>(x, xh, n);
    }
    k_gap_f32<<<B_ * C1_, 256, 0, stream>>>(x, gap1);
    k_attn<<<1, 32, 0, stream>>>(gap1, a1w1, a1b1, a1w2, a1b2, attn1, C1_, C1_ / 4);
    {
        long long tot = (long long)B_ * CO_ * C1_;
        k_mix_irfft<<<(unsigned)((tot + 255) / 256), 256, 0, stream>>>(w1_fr, w1_fi, attn1, wd1, C1_);
    }
    dim3 gconv(2, H_, B_);
    k_conv_wmma<<<gconv, 256, 0, stream>>>(xh, wd1, b1, y1h, nullptr, C1_);

    // ---- layer 2 ----
    k_gap_bf16<<<B_ * C2_, 256, 0, stream>>>(y1h, gap2);
    k_attn<<<1, 32, 0, stream>>>(gap2, a2w1, a2b1, a2w2, a2b2, attn2, C2_, C2_ / 4);
    {
        long long tot = (long long)B_ * CO_ * C2_;
        k_mix_irfft<<<(unsigned)((tot + 255) / 256), 256, 0, stream>>>(w2_fr, w2_fi, attn2, wd2, C2_);
    }
    k_conv_wmma<<<gconv, 256, 0, stream>>>(y1h, wd2, b2, nullptr, (float*)d_out, C2_);
}
